// SparseConv3dMod_54958401519687
// MI455X (gfx1250) — matrix operand order repaired and checked by reading, not compiled, not run on hardware
//
#include <hip/hip_runtime.h>
#include <hip/hip_bf16.h>

// ---------------- problem constants ----------------
constexpr int NB   = 2;            // batch
constexpr int ZD   = 32;           // depth
constexpr int YH   = 256;          // height
constexpr int XW   = 256;          // width
constexpr int NPTS = 150000;       // active points (multiple of 16 -> 9375 tiles)
constexpr int CIN  = 16;
constexpr int CMID = 32;
constexpr int COUT = 64;
constexpr int DO_  = 16;           // (32+2-3)/2+1
constexpr int HO_  = 128;
constexpr int WO_  = 128;
constexpr int GSZ  = NB*ZD*YH*XW;  // 4,194,304 grid cells
constexpr int MOUT = NB*DO_*HO_*WO_; // 524,288 output sites
constexpr int NTILE = NPTS/16;     // 9375
constexpr int RB   = 240;          // reduction blocks
constexpr float EPS = 1e-5f;

typedef __attribute__((ext_vector_type(16))) _Float16 v16h;
typedef __attribute__((ext_vector_type(8)))  _Float16 v8h;
typedef __attribute__((ext_vector_type(8)))  float    v8f;

__device__ __forceinline__ v8f zf() {
  v8f r;
#pragma unroll
  for (int i = 0; i < 8; ++i) r[i] = 0.f;
  return r;
}
__device__ __forceinline__ v8h zh() {
  v8h r;
#pragma unroll
  for (int i = 0; i < 8; ++i) r[i] = (_Float16)0;
  return r;
}
__device__ __forceinline__ v16h cat16(v8h lo, v8h hi) {
  v16h r;
#pragma unroll
  for (int i = 0; i < 8; ++i) { r[i] = lo[i]; r[i+8] = hi[i]; }
  return r;
}
__device__ __forceinline__ v8f wmma_f16(v16h a, v16h b, v8f c) {
  return __builtin_amdgcn_wmma_f32_16x16x32_f16(false, a, false, b, (short)0, c, false, false);
}

// ---------------- utility kernels ----------------
__global__ void fill_i32(int* p, int v, int n) {
  int i = blockIdx.x * blockDim.x + threadIdx.x; if (i < n) p[i] = v;
}
__global__ void fill_f32(float* p, float v, int n) {
  int i = blockIdx.x * blockDim.x + threadIdx.x; if (i < n) p[i] = v;
}
__global__ void cvt_feats(const float* f, _Float16* h, int n) {
  int i = blockIdx.x * blockDim.x + threadIdx.x; if (i < n) h[i] = (_Float16)f[i];
}
__global__ void scatter_grid(const int* coords, int* grid) {
  int i = blockIdx.x * blockDim.x + threadIdx.x; if (i >= NPTS) return;
  const int* c = coords + 4*i;
  int lin = ((c[0]*ZD + c[1])*YH + c[2])*XW + c[3];
  grid[lin] = i;
}
__global__ void build_nbr(const int* coords, const int* grid, int* nbrT) {
  int n = blockIdx.x * blockDim.x + threadIdx.x; int ki = blockIdx.y;
  if (n >= NPTS) return;
  const int* c = coords + 4*n;
  int nz = c[1] + (ki/9) - 1, ny = c[2] + ((ki/3)%3) - 1, nx = c[3] + (ki%3) - 1;
  int v = -1;
  if (nz >= 0 && nz < ZD && ny >= 0 && ny < YH && nx >= 0 && nx < XW)
    v = grid[((c[0]*ZD + nz)*YH + ny)*XW + nx];
  nbrT[ki*NPTS + n] = v;
}

// ---------------- weight packing into WMMA B-fragment layout ----------------
// B fragment (16-bit, KxN=32x16): lane<16 holds col=lane, K {0..7,16..23};
// lane>=16 holds col=lane-16, K {8..15,24..31}.
__global__ void pack_w1(const float* W1, _Float16* w1p) {       // 28*2*32*8
  int id = blockIdx.x * blockDim.x + threadIdx.x;
  if (id >= 28*2*32*8) return;
  int j = id & 7, L = (id >> 3) & 31, t = (id >> 8) & 1, ki = id >> 9;
  _Float16 v = (_Float16)0;
  if (ki < 27) {
    int k = (L < 16) ? j : 8 + j;                // per-offset K=16
    int n = 16*t + (L & 15);
    v = (_Float16)W1[(ki*CIN + k)*CMID + n];
  }
  w1p[id] = v;
}
__global__ void pack_w2(const float* W2, _Float16* w2p) {       // 27*2*32*16
  int id = blockIdx.x * blockDim.x + threadIdx.x;
  if (id >= 27*2*32*16) return;
  int j = id & 15, L = (id >> 4) & 31, t = (id >> 9) & 1, ki = id >> 10;
  int jj = j & 7;
  int k = ((j < 8) ? 0 : 16) + ((L < 16) ? jj : 8 + jj);
  int n = 16*t + (L & 15);
  w2p[id] = (_Float16)W2[(ki*CMID + k)*CMID + n];
}
__global__ void pack_w3(const float* W3, _Float16* w3p) {       // 27*4*32*16
  int id = blockIdx.x * blockDim.x + threadIdx.x;
  if (id >= 27*4*32*16) return;
  int j = id & 15, L = (id >> 4) & 31, t = (id >> 9) & 3, ki = id >> 11;
  int jj = j & 7;
  int k = ((j < 8) ? 0 : 16) + ((L < 16) ? jj : 8 + jj);
  int n = 16*t + (L & 15);
  w3p[id] = (_Float16)W3[(ki*CMID + k)*COUT + n];
}

// ---------------- layer 1: subm conv 16->32 (WMMA, 2 offsets per K-chunk) ----
__global__ __launch_bounds__(256)
void conv1(const _Float16* f16, const int* nbrT, const _Float16* w1p, float* y) {
  int lane = threadIdx.x & 31, wv = threadIdx.x >> 5;
  int tile = blockIdx.x * 8 + wv;
  if (tile >= NTILE) return;                      // wave-uniform
  int base = tile * 16, hi = lane >> 4;
  int p = base + (lane & 15);
  v8f c0 = zf(), c1 = zf();
  for (int kk = 0; kk < 14; ++kk) {
    int ka = 2*kk, kb = 2*kk + 1;
    int na = nbrT[ka*NPTS + p];
    int nb = (kb < 27) ? nbrT[kb*NPTS + p] : -1;
    v8h alo = (na >= 0) ? *(const v8h*)(f16 + na*CIN + hi*8) : zh();
    v8h ahi = (nb >= 0) ? *(const v8h*)(f16 + nb*CIN + hi*8) : zh();
    v16h a = cat16(alo, ahi);
    const _Float16* ba = w1p + ((ka*2 + 0)*32 + lane)*8;
    const _Float16* bb = w1p + ((kb*2 + 0)*32 + lane)*8;   // kb==27 slot is zeros
    v16h b0 = cat16(*(const v8h*)ba, *(const v8h*)bb);
    v16h b1 = cat16(*(const v8h*)(ba + 32*8), *(const v8h*)(bb + 32*8)); // t=1 plane
    c0 = wmma_f16(a, b0, c0);
    c1 = wmma_f16(a, b1, c1);
  }
  int colb = lane & 15;
#pragma unroll
  for (int v = 0; v < 8; ++v) {
    float* yr = y + (size_t)(base + hi*8 + v) * CMID;
    yr[colb]      = c0[v];
    yr[16 + colb] = c1[v];
  }
}

// ---------------- layer 2: subm conv 32->32 (WMMA, 1 offset per K-chunk) -----
__global__ __launch_bounds__(256)
void conv2(const _Float16* x1h, const int* nbrT, const _Float16* w2p, float* y) {
  int lane = threadIdx.x & 31, wv = threadIdx.x >> 5;
  int tile = blockIdx.x * 8 + wv;
  if (tile >= NTILE) return;
  int base = tile * 16, hi = lane >> 4;
  int p = base + (lane & 15);
  v8f c0 = zf(), c1 = zf();
  for (int ki = 0; ki < 27; ++ki) {
    int nn = nbrT[ki*NPTS + p];
    v16h a;
    if (nn >= 0) {
      const _Float16* ap = x1h + nn*CMID + hi*8;
      a = cat16(*(const v8h*)ap, *(const v8h*)(ap + 16));
    } else {
      a = cat16(zh(), zh());
    }
    const _Float16* b0p = w2p + ((ki*2 + 0)*32 + lane)*16;
    const _Float16* b1p = w2p + ((ki*2 + 1)*32 + lane)*16;
    v16h b0 = cat16(*(const v8h*)b0p, *(const v8h*)(b0p + 8));
    v16h b1 = cat16(*(const v8h*)b1p, *(const v8h*)(b1p + 8));
    c0 = wmma_f16(a, b0, c0);
    c1 = wmma_f16(a, b1, c1);
  }
  int colb = lane & 15;
#pragma unroll
  for (int v = 0; v < 8; ++v) {
    float* yr = y + (size_t)(base + hi*8 + v) * CMID;
    yr[colb]      = c0[v];
    yr[16 + colb] = c1[v];
  }
}

// ---------------- layer 3: strided scatter conv 32->64 (WMMA + atomics) -----
__global__ __launch_bounds__(256)
void conv3(const _Float16* x2h, const int* coords, const _Float16* w3p,
           float* out, int* cnt) {
  int lane = threadIdx.x & 31, wv = threadIdx.x >> 5;
  int tile = blockIdx.x * 8 + wv;
  if (tile >= NTILE) return;
  int base = tile * 16, hi = lane >> 4;
  int p = base + (lane & 15);
  const _Float16* ap = x2h + p*CMID + hi*8;
  v16h a = cat16(*(const v8h*)ap, *(const v8h*)(ap + 16));
  const int* cc = coords + 4*p;
  int b = cc[0], z = cc[1], yy = cc[2], xx = cc[3];
  int colb = lane & 15;
  for (int ki = 0; ki < 27; ++ki) {
    int oz = z + 1 - (ki/9), oy = yy + 1 - ((ki/3)%3), ox = xx + 1 - (ki%3);
    bool val = (oz >= 0) && (oy >= 0) && (ox >= 0) &&
               !(oz & 1) && !(oy & 1) && !(ox & 1) &&
               (oz >> 1) < DO_ && (oy >> 1) < HO_ && (ox >> 1) < WO_;
    int lin = val ? (((b*DO_ + (oz >> 1))*HO_ + (oy >> 1))*WO_ + (ox >> 1)) : -1;
    if (lane < 16 && lin >= 0) atomicAdd(cnt + lin, 1);
    const _Float16* bp = w3p + ((ki*4 + 0)*32 + lane)*16;
    v16h b0 = cat16(*(const v8h*)(bp + 0*512),  *(const v8h*)(bp + 0*512 + 8));
    v16h b1 = cat16(*(const v8h*)(bp + 1*512),  *(const v8h*)(bp + 1*512 + 8));
    v16h b2 = cat16(*(const v8h*)(bp + 2*512),  *(const v8h*)(bp + 2*512 + 8));
    v16h b3 = cat16(*(const v8h*)(bp + 3*512),  *(const v8h*)(bp + 3*512 + 8));
    v8f c0 = wmma_f16(a, b0, zf());
    v8f c1 = wmma_f16(a, b1, zf());
    v8f c2 = wmma_f16(a, b2, zf());
    v8f c3 = wmma_f16(a, b3, zf());
#pragma unroll
    for (int v = 0; v < 8; ++v) {
      int lr = __shfl(lin, hi*8 + v, 32);
      if (lr >= 0) {
        float* o = out + (size_t)lr * COUT;
        atomicAdd(o + 0*16 + colb, c0[v]);
        atomicAdd(o + 1*16 + colb, c1[v]);
        atomicAdd(o + 2*16 + colb, c2[v]);
        atomicAdd(o + 3*16 + colb, c3[v]);
      }
    }
  }
}

// ---------------- BN reductions (deterministic two-pass) --------------------
__global__ void reduce32(const float* y, float* part) {   // grid=RB, block=32
  int c = threadIdx.x; float s = 0.f, q = 0.f;
  for (int r = blockIdx.x; r < NPTS; r += gridDim.x) {
    float v = y[(size_t)r*CMID + c]; s += v; q += v*v;
  }
  part[(blockIdx.x*32 + c)*2]     = s;
  part[(blockIdx.x*32 + c)*2 + 1] = q;
}
__global__ void finalize32(const float* part, float* stats) {  // 1x32
  int c = threadIdx.x; float s = 0.f, q = 0.f;
  for (int b = 0; b < RB; ++b) { s += part[(b*32 + c)*2]; q += part[(b*32 + c)*2 + 1]; }
  float mean = s / (float)NPTS;
  float var  = q / (float)NPTS - mean*mean;
  stats[c*2]     = mean;
  stats[c*2 + 1] = rsqrtf(var + EPS);
}
__global__ void bnrelu32(const float* y, const float* stats, const float* g,
                         const float* be, _Float16* xh) {
  int i = blockIdx.x * blockDim.x + threadIdx.x;
  if (i >= NPTS*CMID) return;
  int c = i & 31;
  float v = (y[i] - stats[c*2]) * stats[c*2 + 1] * g[c] + be[c];
  xh[i] = (_Float16)fmaxf(v, 0.f);
}
__global__ void reduce64(const float* out, const int* cnt, float* part) { // grid=RB, block=64
  int c = threadIdx.x; float s = 0.f, q = 0.f, na = 0.f;
  for (int r = blockIdx.x; r < MOUT; r += gridDim.x) {
    if (cnt[r] > 0) {
      float v = out[(size_t)r*COUT + c]; s += v; q += v*v;
      if (c == 0) na += 1.f;
    }
  }
  part[(blockIdx.x*64 + c)*2]     = s;
  part[(blockIdx.x*64 + c)*2 + 1] = q;
  if (c == 0) part[RB*128 + blockIdx.x] = na;
}
__global__ void finalize64(const float* part, float* stats) {  // 1x64
  int c = threadIdx.x; float s = 0.f, q = 0.f, na = 0.f;
  for (int b = 0; b < RB; ++b) {
    s += part[(b*64 + c)*2]; q += part[(b*64 + c)*2 + 1]; na += part[RB*128 + b];
  }
  na = fmaxf(na, 1.f);
  float mean = s / na;
  float var  = q / na - mean*mean;
  stats[c*2]     = mean;
  stats[c*2 + 1] = rsqrtf(var + EPS);
}
__global__ void bnrelu_out(float* out, const int* cnt, const float* stats,
                           const float* g, const float* be) {
  int i = blockIdx.x * blockDim.x + threadIdx.x;
  if (i >= MOUT*COUT) return;
  int r = i >> 6, c = i & 63;
  float v = 0.f;
  if (cnt[r] > 0)
    v = fmaxf((out[i] - stats[c*2]) * stats[c*2 + 1] * g[c] + be[c], 0.f);
  out[i] = v;
}

// ---------------- host side ----------------
static inline size_t al256(size_t x) { return (x + 255) & ~(size_t)255; }

extern "C" void kernel_launch(void* const* d_in, const int* in_sizes, int n_in,
                              void* d_out, int out_size, void* d_ws, size_t ws_size,
                              hipStream_t stream) {
  const float* feats  = (const float*)d_in[0];
  const int*   coords = (const int*)d_in[1];
  const float* W1 = (const float*)d_in[2];
  const float* g1 = (const float*)d_in[4];
  const float* be1 = (const float*)d_in[5];
  const float* W2 = (const float*)d_in[6];
  const float* g2 = (const float*)d_in[8];
  const float* be2 = (const float*)d_in[9];
  const float* W3 = (const float*)d_in[10];
  const float* g3 = (const float*)d_in[12];
  const float* be3 = (const float*)d_in[13];
  float* out = (float*)d_out;   // M x 64 f32, used as scatter accumulator

  char* w = (char*)d_ws; size_t off = 0;
  auto take = [&](size_t bytes) { void* p = w + off; off = al256(off + bytes); return p; };
  int*       grid   = (int*)      take((size_t)GSZ * 4);
  int*       nbrT   = (int*)      take((size_t)27 * NPTS * 4);
  _Float16*  f16    = (_Float16*) take((size_t)NPTS * CIN * 2);
  _Float16*  x1h    = (_Float16*) take((size_t)NPTS * CMID * 2);
  _Float16*  x2h    = (_Float16*) take((size_t)NPTS * CMID * 2);
  float*     yb     = (float*)    take((size_t)NPTS * CMID * 4);
  _Float16*  w1p    = (_Float16*) take((size_t)28*2*32*8 * 2);
  _Float16*  w2p    = (_Float16*) take((size_t)27*2*32*16 * 2);
  _Float16*  w3p    = (_Float16*) take((size_t)27*4*32*16 * 2);
  int*       cnt    = (int*)      take((size_t)MOUT * 4);
  float*     part   = (float*)    take((size_t)RB * 64 * 4);
  float*     part3  = (float*)    take((size_t)(RB*128 + RB) * 4);
  float*     stats1 = (float*)    take(64 * 4);
  float*     stats2 = (float*)    take(64 * 4);
  float*     stats3 = (float*)    take(128 * 4);

  // init
  fill_i32<<<(GSZ + 255)/256, 256, 0, stream>>>(grid, -1, GSZ);
  fill_i32<<<(MOUT + 255)/256, 256, 0, stream>>>(cnt, 0, MOUT);
  fill_f32<<<(MOUT*COUT + 255)/256, 256, 0, stream>>>(out, 0.f, MOUT*COUT);

  // preprocessing
  cvt_feats<<<(NPTS*CIN + 255)/256, 256, 0, stream>>>(feats, f16, NPTS*CIN);
  pack_w1<<<(28*2*32*8 + 255)/256, 256, 0, stream>>>(W1, w1p);
  pack_w2<<<(27*2*32*16 + 255)/256, 256, 0, stream>>>(W2, w2p);
  pack_w3<<<(27*4*32*16 + 255)/256, 256, 0, stream>>>(W3, w3p);
  scatter_grid<<<(NPTS + 255)/256, 256, 0, stream>>>(coords, grid);
  {
    dim3 g((NPTS + 255)/256, 27);
    build_nbr<<<g, 256, 0, stream>>>(coords, grid, nbrT);
  }

  int convBlocks = (NTILE + 7) / 8;

  // layer 1
  conv1<<<convBlocks, 256, 0, stream>>>(f16, nbrT, w1p, yb);
  reduce32<<<RB, 32, 0, stream>>>(yb, part);
  finalize32<<<1, 32, 0, stream>>>(part, stats1);
  bnrelu32<<<(NPTS*CMID + 255)/256, 256, 0, stream>>>(yb, stats1, g1, be1, x1h);

  // layer 2
  conv2<<<convBlocks, 256, 0, stream>>>(x1h, nbrT, w2p, yb);
  reduce32<<<RB, 32, 0, stream>>>(yb, part);
  finalize32<<<1, 32, 0, stream>>>(part, stats2);
  bnrelu32<<<(NPTS*CMID + 255)/256, 256, 0, stream>>>(yb, stats2, g2, be2, x2h);

  // layer 3 (scatter) + masked BN-ReLU
  conv3<<<convBlocks, 256, 0, stream>>>(x2h, coords, w3p, out, cnt);
  reduce64<<<RB, 64, 0, stream>>>(out, cnt, part3);
  finalize64<<<1, 64, 0, stream>>>(part3, stats3);
  bnrelu_out<<<(MOUT*COUT + 255)/256, 256, 0, stream>>>(out, cnt, stats3, g3, be3);

  (void)in_sizes; (void)n_in; (void)out_size; (void)ws_size;
}
